// DGDNN_16733192585659
// MI455X (gfx1250) — matrix operand order repaired and checked by reading, not compiled, not run on hardware
//
#include <hip/hip_runtime.h>
#include <hip/hip_bf16.h>
#include <cmath>

// DGDNN forward for MI455X (gfx1250, wave32, WMMA f32_16x16x32_f16 + async LDS).
// B=32, N=1024 nodes, F_IN=64, HID=RAW=OUT=D1=D2=128, H=2 heads (dh=64), CLS=2.

#define BB   32
#define NN   1024
#define FIN  64
#define HIDD 128
#define BN   (BB * NN)   // 32768 rows

typedef __attribute__((ext_vector_type(16))) _Float16 v16h;
typedef __attribute__((ext_vector_type(8)))  _Float16 v8h;
typedef __attribute__((ext_vector_type(2)))  _Float16 v2h;
typedef __attribute__((ext_vector_type(8)))  float    v8f;
typedef __attribute__((ext_vector_type(4)))  float    v4f;

__device__ __forceinline__ v8f wmma_f16(v16h a, v16h b, v8f c) {
  // D = A(16x32 f16) * B(32x16 f16) + C(16x16 f32)
  return __builtin_amdgcn_wmma_f32_16x16x32_f16(false, a, false, b, (short)0, c,
                                                false, false);
}

// 32-bit LDS offset of a shared-memory object (for async-to-LDS VDST).
__device__ __forceinline__ unsigned lds_addr_of(const void* p) {
  return (unsigned)(unsigned long long)(__attribute__((address_space(3))) const void*)p;
}

// ---- fragment loaders (wave32, contiguous k order shared by A and B), vectorized ----
// A fragment: 16x32 f16 row-major tile, leading dim `ld` (multiple of 8, 16B base).
// lane L: row m=L&15; element e -> k = e + 16*(L>>4)  (contiguous along k)
__device__ __forceinline__ v16h load_a_frag_lds(const _Float16* base, int ld) {
  const int lane = threadIdx.x & 31;
  const int m  = lane & 15;
  const int kb = (lane >> 4) * 16;
  const v8h lo = *(const v8h*)(base + m * ld + kb);
  const v8h hi = *(const v8h*)(base + m * ld + kb + 8);
  return __builtin_shufflevector(lo, hi, 0, 1, 2, 3, 4, 5, 6, 7,
                                         8, 9, 10, 11, 12, 13, 14, 15);
}

// B fragment from a TRANSPOSED tile sBT[col][k] (32 k x 16 cols logical).
// lane L: col n=L&15; element e -> k = e + 16*(L>>4)  (contiguous along k)
__device__ __forceinline__ v16h load_bT_frag_lds(const _Float16* base, int ld) {
  const int lane = threadIdx.x & 31;
  const int n  = lane & 15;
  const int kb = (lane >> 4) * 16;
  const v8h lo = *(const v8h*)(base + n * ld + kb);
  const v8h hi = *(const v8h*)(base + n * ld + kb + 8);
  return __builtin_shufflevector(lo, hi, 0, 1, 2, 3, 4, 5, 6, 7,
                                         8, 9, 10, 11, 12, 13, 14, 15);
}

// Convert 16 consecutive f32 (16B-aligned) to v16h via 4x b128 loads.
__device__ __forceinline__ v16h cvt_f32x16_f16(const float* p) {
  const v4f* q = (const v4f*)p;
  const v4f a = q[0], b = q[1], c = q[2], d = q[3];
  v16h r;
#pragma unroll
  for (int i = 0; i < 4; ++i) {
    r[i]      = (_Float16)a[i];
    r[4 + i]  = (_Float16)b[i];
    r[8 + i]  = (_Float16)c[i];
    r[12 + i] = (_Float16)d[i];
  }
  return r;
}

// =============================================================================
// Kernel 1: theta softmax + Q[l] = sum_k theta_soft[l,k] * T[l,k]   (K=3)
// =============================================================================
__global__ __launch_bounds__(256) void compute_q_kernel(
    const float* __restrict__ T, const float* __restrict__ theta,
    float* __restrict__ Qout) {
  const size_t idx = (size_t)blockIdx.x * blockDim.x + threadIdx.x;
  const size_t nn = (size_t)NN * NN;
  if (idx >= 2 * nn) return;
  const int    l   = (int)(idx / nn);
  const size_t off = idx - (size_t)l * nn;
  const float t0 = theta[l * 3 + 0], t1 = theta[l * 3 + 1], t2 = theta[l * 3 + 2];
  const float mx = fmaxf(t0, fmaxf(t1, t2));
  const float e0 = __expf(t0 - mx), e1 = __expf(t1 - mx), e2 = __expf(t2 - mx);
  const float inv = 1.0f / (e0 + e1 + e2);
  Qout[idx] = inv * (e0 * T[(size_t)(l * 3 + 0) * nn + off] +
                     e1 * T[(size_t)(l * 3 + 1) * nn + off] +
                     e2 * T[(size_t)(l * 3 + 2) * nn + off]);
}

// =============================================================================
// Kernel 2: generic WMMA GEMM  out[M,128] = act( [src0 | src1] @ W + bias ) (+res)
//   Block = 128 thr = 4 waves, 64 rows. Weight tile staged transposed.
// =============================================================================
template <int K0, int K1, bool RELU, bool RESID>
__global__ __launch_bounds__(128) void gemm128_kernel(
    const float* __restrict__ src0, const float* __restrict__ src1,
    const float* __restrict__ W, const float* __restrict__ bias,
    const float* __restrict__ resid, float* __restrict__ out) {
  constexpr int KD = K0 + K1;
  __shared__ __align__(16) _Float16 sA[64][40];     // [row][k]
  __shared__ __align__(16) _Float16 sBT[128][40];   // [col][k] (transposed)

  const int w    = threadIdx.x >> 5;
  const int lane = threadIdx.x & 31;
  const int tid  = threadIdx.x;
  const size_t rowBase = (size_t)blockIdx.x * 64;

  v8f acc[8];
#pragma unroll
  for (int i = 0; i < 8; ++i) acc[i] = {};

  for (int kc = 0; kc < KD; kc += 32) {
    // stage activations, packed f16 pairs (virtual concat of src0|src1)
#pragma unroll
    for (int j = 0; j < 8; ++j) {
      const int li = j * 128 + tid;        // 0..1023 pairs
      const int r  = li >> 4;              // 0..63
      const int kk = (li & 15) * 2;        // even k
      const int kg = kc + kk;
      const size_t gr = rowBase + r;
      float x0, x1;
      if constexpr (K1 > 0) {
        const float* p = (kg < K0) ? (src0 + gr * (size_t)K0 + kg)
                                   : (src1 + gr * (size_t)K1 + (kg - K0));
        x0 = p[0]; x1 = p[1];
      } else {
        const float* p = src0 + gr * (size_t)K0 + kg;
        x0 = p[0]; x1 = p[1];
      }
      v2h t = {(_Float16)x0, (_Float16)x1};
      *(v2h*)&sA[r][kk] = t;
    }
    // stage weights transposed, packed pairs along k
#pragma unroll
    for (int j = 0; j < 16; ++j) {
      const int li = j * 128 + tid;        // 0..2047 pairs
      const int c  = li & 127;
      const int kk = (li >> 7) * 2;
      v2h t = {(_Float16)W[(size_t)(kc + kk) * 128 + c],
               (_Float16)W[(size_t)(kc + kk + 1) * 128 + c]};
      *(v2h*)&sBT[c][kk] = t;
    }
    __syncthreads();
    const v16h a = load_a_frag_lds(&sA[w * 16][0], 40);
#pragma unroll
    for (int nt = 0; nt < 8; ++nt) {
      const v16h bf = load_bT_frag_lds(&sBT[nt * 16][0], 40);
      acc[nt] = wmma_f16(a, bf, acc[nt]);
    }
    __syncthreads();
  }

  // epilogue; C layout: row = e + 8*(lane>>4), col = lane&15
  const int n16 = lane & 15;
  const int rb  = (lane >> 4) * 8;
#pragma unroll
  for (int nt = 0; nt < 8; ++nt) {
    const int col = nt * 16 + n16;
    const float bv = bias[col];
#pragma unroll
    for (int e = 0; e < 8; ++e) {
      const size_t gr = rowBase + w * 16 + rb + e;
      float v = acc[nt][e] + bv;
      if constexpr (RELU)  v = fmaxf(v, 0.0f);
      if constexpr (RESID) v += resid[gr * 128 + col];
      out[gr * 128 + col] = v;
    }
  }
}

// =============================================================================
// Kernel 3: masked diffusion  Z[b] = (Q ⊙ A[b]) @ Hin[b]   (K = N = 1024)
// =============================================================================
template <int F>
__global__ __launch_bounds__(128) void diffuse_kernel(
    const float* __restrict__ Qd, const float* __restrict__ A,
    const float* __restrict__ Hin, float* __restrict__ Z) {
  __shared__ __align__(16) _Float16 sS[64][40];     // masked transition [row][k]
  __shared__ __align__(16) _Float16 sHT[F][40];     // features transposed [col][k]

  const int w    = threadIdx.x >> 5;
  const int lane = threadIdx.x & 31;
  const int tid  = threadIdx.x;
  const int b    = blockIdx.y;
  const size_t rowBase = (size_t)blockIdx.x * 64;
  const float* Ab = A   + (size_t)b * NN * NN;
  const float* Hb = Hin + (size_t)b * NN * F;

  constexpr int NT   = F / 16;
  constexpr int LOGF = (F == 128) ? 7 : 6;
  v8f acc[NT];
#pragma unroll
  for (int i = 0; i < NT; ++i) acc[i] = {};

  for (int kc = 0; kc < NN; kc += 32) {
    // masked transition tile S = Q * A, packed pairs (fused mask; one pass over A)
#pragma unroll
    for (int j = 0; j < 8; ++j) {
      const int li = j * 128 + tid;
      const int r  = li >> 4;
      const int kk = (li & 15) * 2;
      const size_t gi = (rowBase + r) * (size_t)NN + (kc + kk);
      const float2 qv = *(const float2*)(Qd + gi);
      const float2 av = *(const float2*)(Ab + gi);
      v2h t = {(_Float16)(qv.x * av.x), (_Float16)(qv.y * av.y)};
      *(v2h*)&sS[r][kk] = t;
    }
    // feature tile transposed, packed pairs along k
#pragma unroll
    for (int j = 0; j < (16 * F) / 128; ++j) {
      const int li = j * 128 + tid;
      const int c  = li & (F - 1);
      const int kk = (li >> LOGF) * 2;
      v2h t = {(_Float16)Hb[(size_t)(kc + kk) * F + c],
               (_Float16)Hb[(size_t)(kc + kk + 1) * F + c]};
      *(v2h*)&sHT[c][kk] = t;
    }
    __syncthreads();
    const v16h a = load_a_frag_lds(&sS[w * 16][0], 40);
#pragma unroll
    for (int nt = 0; nt < NT; ++nt) {
      const v16h bf = load_bT_frag_lds(&sHT[nt * 16][0], 40);
      acc[nt] = wmma_f16(a, bf, acc[nt]);
    }
    __syncthreads();
  }

  const int n16 = lane & 15;
  const int rb  = (lane >> 4) * 8;
#pragma unroll
  for (int nt = 0; nt < NT; ++nt) {
#pragma unroll
    for (int e = 0; e < 8; ++e) {
      const size_t gr = rowBase + w * 16 + rb + e;
      Z[((size_t)b * NN + gr) * F + nt * 16 + n16] = acc[nt][e];
    }
  }
}

// =============================================================================
// Kernel 4: flash attention. Block = 4 waves (same b,h; 4 row-tiles); the
// 32-key K/V chunks are staged block-wide in LDS with gfx1250 async-to-LDS
// copies (ASYNCcnt), then consumed by WMMA with online softmax.
// =============================================================================
__global__ __launch_bounds__(128) void flash_attn_kernel(
    const float* __restrict__ Qm, const float* __restrict__ Km,
    const float* __restrict__ Vm, float* __restrict__ Ctx) {
  __shared__ __align__(16) float    sK[32][72];     // keys x dh (f32, padded)
  __shared__ __align__(16) float    sV[32][72];
  __shared__ __align__(16) float    sSc[4][32][20]; // per-wave scores [col][row]
  __shared__ __align__(16) _Float16 sP[4][16][40];  // per-wave probs [row][k]
  __shared__ float sStat[4][16];

  const int w    = threadIdx.x >> 5;
  const int lane = threadIdx.x & 31;
  const int tile = blockIdx.x * 4 + w;   // 0..63 row tiles
  const int h    = blockIdx.y;
  const int b    = blockIdx.z;

  const size_t baseBN = (size_t)b * NN * HIDD + (size_t)h * 64;
  const unsigned ldsK = lds_addr_of(&sK[0][0]);
  const unsigned ldsV = lds_addr_of(&sV[0][0]);

  // Q fragments (16 rows x dh=64), pre-scaled by 1/sqrt(64)
  const int m   = lane & 15;
  const int kbA = (lane >> 4) * 16;
  const float* qrow = Qm + baseBN + (size_t)(tile * 16 + m) * HIDD;
  v16h qa0, qa1;
#pragma unroll
  for (int e = 0; e < 8; ++e) {
    qa0[e]     = (_Float16)(qrow[kbA + e]      * 0.125f);
    qa0[e + 8] = (_Float16)(qrow[8 + kbA + e] * 0.125f);
    qa1[e]     = (_Float16)(qrow[32 + kbA + e] * 0.125f);
    qa1[e + 8] = (_Float16)(qrow[40 + kbA + e] * 0.125f);
  }

  v8f o0 = {}, o1 = {}, o2 = {}, o3 = {};
  float m_run = -3.0e38f, l_run = 0.0f;   // valid in lanes 0..15 (row = lane)

  const int n16 = lane & 15;
  const int kbB = (lane >> 4) * 16;
  const int rbS = (lane >> 4) * 8;

  for (int j0 = 0; j0 < NN; j0 += 32) {
    __syncthreads();  // previous iteration's LDS reads finished before restage

    // ---- async staging of K/V chunk: 8KB each, 16B per lane per op ----
#pragma unroll
    for (int ii = 0; ii < 4; ++ii) {
      const int ci = (w * 4 + ii) * 32 + lane;   // 0..511 16B-chunks
      const int r  = ci >> 4;                    // key row 0..31
      const int ch = ci & 15;                    // 16B chunk within 64-float row
      const unsigned dst = (unsigned)(r * (72 * 4) + ch * 16);
      const unsigned long long gk =
          (unsigned long long)(const void*)(Km + baseBN + (size_t)(j0 + r) * HIDD) +
          (unsigned long long)(ch * 16);
      const unsigned long long gv =
          (unsigned long long)(const void*)(Vm + baseBN + (size_t)(j0 + r) * HIDD) +
          (unsigned long long)(ch * 16);
      asm volatile("global_load_async_to_lds_b128 %0, %1, off"
                   :: "v"(ldsK + dst), "v"(gk) : "memory");
      asm volatile("global_load_async_to_lds_b128 %0, %1, off"
                   :: "v"(ldsV + dst), "v"(gv) : "memory");
    }
    asm volatile("s_wait_asynccnt 0" ::: "memory");
    __syncthreads();

    // ---- scores: two 16-key tiles via WMMA (contraction over dh) ----
#pragma unroll
    for (int t = 0; t < 2; ++t) {
      const int key = t * 16 + n16;
      const v16h k0 = cvt_f32x16_f16(&sK[key][kbB]);        // dh 0..31 slice
      const v16h k1 = cvt_f32x16_f16(&sK[key][32 + kbB]);   // dh 32..63 slice
      v8f s = {};
      s = wmma_f16(qa0, k0, s);
      s = wmma_f16(qa1, k1, s);
      // store transposed -> two b128 stores per lane
      v4f lo = {s[0], s[1], s[2], s[3]};
      v4f hi = {s[4], s[5], s[6], s[7]};
      *(v4f*)&sSc[w][key][rbS]     = lo;
      *(v4f*)&sSc[w][key][rbS + 4] = hi;
    }
    asm volatile("s_wait_dscnt 0" ::: "memory");

    // ---- online softmax stats: lanes 0..15 own one row each ----
    if (lane < 16) {
      float sc[32];
#pragma unroll
      for (int c = 0; c < 32; ++c) sc[c] = sSc[w][c][lane];
      float mt = -3.0e38f;
#pragma unroll
      for (int c = 0; c < 32; ++c) mt = fmaxf(mt, sc[c]);
      const float m_new = fmaxf(m_run, mt);
      const float alpha = __expf(m_run - m_new);
      float lsum = 0.0f;
#pragma unroll
      for (int c = 0; c < 32; c += 2) {
        const float p0 = __expf(sc[c] - m_new);
        const float p1 = __expf(sc[c + 1] - m_new);
        lsum += p0 + p1;
        v2h pp = {(_Float16)p0, (_Float16)p1};
        *(v2h*)&sP[w][lane][c] = pp;
      }
      l_run = l_run * alpha + lsum;
      m_run = m_new;
      sStat[w][lane] = alpha;
    }
    asm volatile("s_wait_dscnt 0" ::: "memory");

    // ---- rescale accumulators by per-row alpha (EXEC all-ones again) ----
    float av[8];
#pragma unroll
    for (int e = 0; e < 8; ++e) av[e] = sStat[w][rbS + e];
#pragma unroll
    for (int e = 0; e < 8; ++e) {
      o0[e] *= av[e]; o1[e] *= av[e]; o2[e] *= av[e]; o3[e] *= av[e];
    }

    // ---- O += P @ V  (contraction over 32 keys) ----
    const v16h pa = load_a_frag_lds(&sP[w][0][0], 40);
#pragma unroll
    for (int ct = 0; ct < 4; ++ct) {
      v16h vf;
#pragma unroll
      for (int e = 0; e < 16; ++e)
        vf[e] = (_Float16)sV[kbB + e][ct * 16 + n16];
      if      (ct == 0) o0 = wmma_f16(pa, vf, o0);
      else if (ct == 1) o1 = wmma_f16(pa, vf, o1);
      else if (ct == 2) o2 = wmma_f16(pa, vf, o2);
      else              o3 = wmma_f16(pa, vf, o3);
    }
  }

  // ---- finalize: divide by l, store ctx ----
  if (lane < 16) sStat[w][lane] = 1.0f / l_run;
  asm volatile("s_wait_dscnt 0" ::: "memory");
#pragma unroll
  for (int e = 0; e < 8; ++e) {
    const float inv = sStat[w][rbS + e];
    float* crow = Ctx + baseBN + (size_t)(tile * 16 + rbS + e) * HIDD;
    crow[ 0 + n16] = o0[e] * inv;
    crow[16 + n16] = o1[e] * inv;
    crow[32 + n16] = o2[e] * inv;
    crow[48 + n16] = o3[e] * inv;
  }
}

// =============================================================================
// Kernel 5: final classifier  out[r,0..1] = hp[r,:] @ W_fin + b_fin
// =============================================================================
__global__ __launch_bounds__(256) void final_head_kernel(
    const float* __restrict__ hp, const float* __restrict__ Wf,
    const float* __restrict__ bf, float* __restrict__ out) {
  const int r = blockIdx.x * blockDim.x + threadIdx.x;
  if (r >= BN) return;
  float a0 = bf[0], a1 = bf[1];
  const float* row = hp + (size_t)r * 128;
#pragma unroll 8
  for (int d = 0; d < 128; ++d) {
    const float x = row[d];
    a0 += x * Wf[d * 2 + 0];
    a1 += x * Wf[d * 2 + 1];
  }
  out[r * 2 + 0] = a0;
  out[r * 2 + 1] = a1;
}

// =============================================================================
extern "C" void kernel_launch(void* const* d_in, const int* in_sizes, int n_in,
                              void* d_out, int out_size, void* d_ws, size_t ws_size,
                              hipStream_t stream) {
  const float* X     = (const float*)d_in[0];
  const float* A     = (const float*)d_in[1];
  const float* T     = (const float*)d_in[2];
  const float* theta = (const float*)d_in[3];
  const float* Wraw  = (const float*)d_in[4];
  const float* braw  = (const float*)d_in[5];
  const float* Wd0   = (const float*)d_in[6];
  const float* bd0   = (const float*)d_in[7];
  const float* Wd1   = (const float*)d_in[8];
  const float* bd1   = (const float*)d_in[9];
  const float* Wfin  = (const float*)d_in[10];
  const float* bfin  = (const float*)d_in[11];
  const float* Wq0 = (const float*)d_in[12]; const float* bq0 = (const float*)d_in[13];
  const float* Wk0 = (const float*)d_in[14]; const float* bk0 = (const float*)d_in[15];
  const float* Wv0 = (const float*)d_in[16]; const float* bv0 = (const float*)d_in[17];
  const float* Wo0 = (const float*)d_in[18]; const float* bo0 = (const float*)d_in[19];
  const float* Wq1 = (const float*)d_in[20]; const float* bq1 = (const float*)d_in[21];
  const float* Wk1 = (const float*)d_in[22]; const float* bk1 = (const float*)d_in[23];
  const float* Wv1 = (const float*)d_in[24]; const float* bv1 = (const float*)d_in[25];
  const float* Wo1 = (const float*)d_in[26]; const float* bo1 = (const float*)d_in[27];
  float* out = (float*)d_out;

  // ---- workspace carve-up (floats); total ~35.7M floats ~= 143 MB ----
  float* ws = (float*)d_ws;
  size_t off = 0;
  const size_t BUF = (size_t)BN * 128;
  float* Qd   = ws + off; off += (size_t)2 * NN * NN;
  float* hpA  = ws + off; off += BUF;   // h_prime (layer0 raw proj / final)
  float* hpB  = ws + off; off += BUF;   // h_prime after layer0 attention
  float* hbuf = ws + off; off += BUF;   // diffused features h
  float* Zb   = ws + off; off += BUF;   // diffusion pre-activation
  float* qb   = ws + off; off += BUF;
  float* kb   = ws + off; off += BUF;
  float* vb   = ws + off; off += BUF;
  float* ctx  = ws + off; off += BUF;
  (void)ws_size; (void)n_in; (void)in_sizes; (void)out_size;

  const dim3 blk(128);
  const dim3 gGemm(BN / 64);              // 512 blocks
  const dim3 gDiff(NN / 64, BB);          // 16 x 32
  const dim3 gAttn(NN / 16 / 4, 2, BB);   // 16 x 2 x 32

  // 0) Q[l] = softmax(theta)[l] . T[l]
  compute_q_kernel<<<(2 * NN * NN + 255) / 256, 256, 0, stream>>>(T, theta, Qd);

  // 1) h_prime = X @ W_raw + b_raw
  gemm128_kernel<64, 0, false, false><<<gGemm, blk, 0, stream>>>(
      X, nullptr, Wraw, braw, nullptr, hpA);

  // 2) layer 0 diffusion: h = relu(((Q0*A) @ X) @ Wd0 + bd0)
  diffuse_kernel<64><<<gDiff, blk, 0, stream>>>(Qd, A, X, Zb);
  gemm128_kernel<64, 0, true, false><<<gGemm, blk, 0, stream>>>(
      Zb, nullptr, Wd0, bd0, nullptr, hbuf);

  // 3) attention 0 on x = [h || h_prime]
  gemm128_kernel<128, 128, false, false><<<gGemm, blk, 0, stream>>>(
      hbuf, hpA, Wq0, bq0, nullptr, qb);
  gemm128_kernel<128, 128, false, false><<<gGemm, blk, 0, stream>>>(
      hbuf, hpA, Wk0, bk0, nullptr, kb);
  gemm128_kernel<128, 128, false, false><<<gGemm, blk, 0, stream>>>(
      hbuf, hpA, Wv0, bv0, nullptr, vb);
  flash_attn_kernel<<<gAttn, blk, 0, stream>>>(qb, kb, vb, ctx);
  gemm128_kernel<128, 0, true, false><<<gGemm, blk, 0, stream>>>(
      ctx, nullptr, Wo0, bo0, nullptr, hpB);

  // 4) layer 1 diffusion: h = relu(((Q1*A) @ h) @ Wd1 + bd1)
  diffuse_kernel<128><<<gDiff, blk, 0, stream>>>(Qd + (size_t)NN * NN, A, hbuf, Zb);
  gemm128_kernel<128, 0, true, false><<<gGemm, blk, 0, stream>>>(
      Zb, nullptr, Wd1, bd1, nullptr, hbuf);

  // 5) attention 1 + residual: hp = hpB + relu(attn(...) @ Wo1 + bo1)
  gemm128_kernel<128, 128, false, false><<<gGemm, blk, 0, stream>>>(
      hbuf, hpB, Wq1, bq1, nullptr, qb);
  gemm128_kernel<128, 128, false, false><<<gGemm, blk, 0, stream>>>(
      hbuf, hpB, Wk1, bk1, nullptr, kb);
  gemm128_kernel<128, 128, false, false><<<gGemm, blk, 0, stream>>>(
      hbuf, hpB, Wv1, bv1, nullptr, vb);
  flash_attn_kernel<<<gAttn, blk, 0, stream>>>(qb, kb, vb, ctx);
  gemm128_kernel<128, 0, true, true><<<gGemm, blk, 0, stream>>>(
      ctx, nullptr, Wo1, bo1, hpB, hpA);

  // 6) classifier
  final_head_kernel<<<(BN + 255) / 256, 256, 0, stream>>>(hpA, Wfin, bfin, out);
}
